// QuadraticAttention_82652350644285
// MI455X (gfx1250) — matrix operand order repaired and checked by reading, not compiled, not run on hardware
//
#include <hip/hip_runtime.h>
#include <hip/hip_bf16.h>

#define NB 4
#define LQ 1024
#define SK 1024
#define EE 512
#define HH 8
#define DH 64

typedef __attribute__((ext_vector_type(8)))  float    v8f;
typedef __attribute__((ext_vector_type(8)))  unsigned v8u;
typedef __attribute__((ext_vector_type(16))) __bf16   v16bf;

#if defined(__gfx1250__) && __has_builtin(__builtin_amdgcn_global_load_async_to_lds_b128)
#define USE_ASYNC_LDS 1
typedef int vec4i __attribute__((vector_size(16)));
typedef __attribute__((address_space(1))) vec4i gv4i;   // global int4
typedef __attribute__((address_space(3))) vec4i lv4i;   // LDS int4
#else
#define USE_ASYNC_LDS 0
#endif

__device__ __forceinline__ unsigned short f2bf(float f) {
  unsigned u = __builtin_bit_cast(unsigned, f);
  u += 0x7fffu + ((u >> 16) & 1u);          // round-to-nearest-even
  return (unsigned short)(u >> 16);
}
__device__ __forceinline__ unsigned packbf(float lo, float hi) {
  return (unsigned)f2bf(lo) | ((unsigned)f2bf(hi) << 16);
}
__device__ __forceinline__ float bf2f(unsigned short b) {
  return __builtin_bit_cast(float, (unsigned)b << 16);
}
__device__ __forceinline__ v16bf frag2(uint4 lo, uint4 hi) {
  v8u u;
  u[0] = lo.x; u[1] = lo.y; u[2] = lo.z; u[3] = lo.w;
  u[4] = hi.x; u[5] = hi.y; u[6] = hi.z; u[7] = hi.w;
  return __builtin_bit_cast(v16bf, u);
}

// ---------------------------------------------------------------------------
// Kernel 0: fp32 -> packed bf16 conversion (done once; GEMMs then load b128)
// ---------------------------------------------------------------------------
__global__ void cvt_bf16(const float* __restrict__ src, unsigned* __restrict__ dst,
                         int ndw) {
  const int i = blockIdx.x * blockDim.x + threadIdx.x;
  if (i < ndw) {
    const float2 f = ((const float2*)src)[i];
    dst[i] = packbf(f.x, f.y);
  }
}

// ---------------------------------------------------------------------------
// Kernel 1: Q/K/V projections from bf16 operands.
// out[n][h][l][d] (bf16) = (X @ W^T), channel e = d*H + h (heads innermost).
// One wave per 16x16 C tile; K=512 as 16 bf16 WMMAs; fragments = 2 b128 loads.
// ---------------------------------------------------------------------------
__global__ void __launch_bounds__(128) proj_qkv(
    const unsigned short* __restrict__ Xq, const unsigned short* __restrict__ Xk,
    const unsigned short* __restrict__ Wqb, const unsigned short* __restrict__ Wkb,
    const unsigned short* __restrict__ Wvb,
    unsigned short* __restrict__ Qbf, unsigned short* __restrict__ Kbf,
    unsigned short* __restrict__ Vbf) {
  const int wid  = blockIdx.x * 4 + (threadIdx.x >> 5);
  const int lane = threadIdx.x & 31;
  const int half = lane >> 4, l15 = lane & 15;
  const int tilesPer = (NB * LQ / 16) * (EE / 16);   // 8192
  const int proj = wid / tilesPer;
  const int rem  = wid % tilesPer;
  const int m0   = (rem / (EE / 16)) * 16;
  const int c0   = (rem % (EE / 16)) * 16;

  const unsigned short* X = (proj == 0) ? Xq : Xk;   // V projects `key` (faithful)
  const unsigned short* W = (proj == 0) ? Wqb : (proj == 1 ? Wkb : Wvb);
  unsigned short* O = (proj == 0) ? Qbf : (proj == 1 ? Kbf : Vbf);

  const uint4* A4 = (const uint4*)X;                 // row stride: 64 uint4 (512 bf16)
  const uint4* B4 = (const uint4*)W;
  const int arow = (m0 + l15) * (EE / 8);
  const int brow = (c0 + l15) * (EE / 8);

  v8f acc = {};
  for (int kc = 0; kc < EE / 32; ++kc) {
    const int e4 = kc * 4;                           // e0/8
    const uint4 alo = A4[arow + e4 + half];
    const uint4 ahi = A4[arow + e4 + 2 + half];
    const uint4 blo = B4[brow + e4 + half];
    const uint4 bhi = B4[brow + e4 + 2 + half];
    acc = __builtin_amdgcn_wmma_f32_16x16x32_bf16(false, frag2(alo, ahi), false,
                                                  frag2(blo, bhi), (short)0, acc,
                                                  false, false);
  }
#pragma unroll
  for (int j = 0; j < 8; ++j) {
    const int m = m0 + (half ? 8 + j : j);
    const int c = c0 + l15;
    const int n = m / LQ, l = m % LQ;
    const int h = c % HH, d = c / HH;
    O[(((size_t)n * HH + h) * LQ + l) * DH + d] = f2bf(acc[j]);
  }
}

// ---------------------------------------------------------------------------
// Kernel 2: row squared-norms of bf16 Q and K (per (n,h,row))
// ---------------------------------------------------------------------------
__global__ void row_norms(const unsigned short* __restrict__ Qbf,
                          const unsigned short* __restrict__ Kbf,
                          float* __restrict__ q2, float* __restrict__ k2) {
  const int r = blockIdx.x * blockDim.x + threadIdx.x;
  if (r >= NB * HH * LQ) return;
  const unsigned* q32 = (const unsigned*)Qbf + (size_t)r * (DH / 2);
  const unsigned* k32 = (const unsigned*)Kbf + (size_t)r * (DH / 2);
  float sq = 0.f, sk = 0.f;
#pragma unroll
  for (int i = 0; i < DH / 2; ++i) {
    unsigned uq = q32[i], uk = k32[i];
    float a = bf2f((unsigned short)uq), b = bf2f((unsigned short)(uq >> 16));
    sq += a * a + b * b;
    a = bf2f((unsigned short)uk); b = bf2f((unsigned short)(uk >> 16));
    sk += a * a + b * b;
  }
  q2[r] = sq;
  k2[r] = sk;
}

// ---------------------------------------------------------------------------
// Kernel 3: fused distance-attention (flash style). Block = 4 waves x 16 rows,
// one (n,h); loop S in 32-key tiles staged in LDS (K via async-to-LDS),
// b128 fragment loads throughout.
// ---------------------------------------------------------------------------
__global__ void __launch_bounds__(128) flash_attn(
    const unsigned short* __restrict__ Qbf, const unsigned short* __restrict__ Kbf,
    const unsigned short* __restrict__ Vbf, const float* __restrict__ q2,
    const float* __restrict__ k2, unsigned short* __restrict__ ctx) {
  __shared__ __align__(16) unsigned short kt[32 * DH];     // K tile [s][d]
  __shared__ __align__(16) unsigned short vt[DH * 32];     // V tile transposed [d][s]
  __shared__ __align__(16) unsigned short pst[4][16 * 32]; // per-wave P staging
  __shared__ float k2s[32];

  const int tid  = threadIdx.x;
  const int w    = tid >> 5, lane = tid & 31;
  const int half = lane >> 4, l15 = lane & 15;
  const int nh   = blockIdx.x / (LQ / 64);
  const int qt   = blockIdx.x % (LQ / 64);
  const int qrow0 = qt * 64 + w * 16;

  const unsigned* K32 = (const unsigned*)(Kbf + (size_t)nh * SK * DH);
  const unsigned* V32 = (const unsigned*)(Vbf + (size_t)nh * SK * DH);
  const float* q2h = q2 + (size_t)nh * LQ;
  const float* k2h = k2 + (size_t)nh * SK;

  // Q A-fragments: rows qrow0+l15, 8 uint4 per row; two 16x32 K-chunks
  const uint4* Q4 = (const uint4*)(Qbf + (size_t)nh * LQ * DH);
  const int qrow4 = (qrow0 + l15) * (DH / 8);
  const v16bf qa0 = frag2(Q4[qrow4 + half], Q4[qrow4 + 2 + half]);
  const v16bf qa1 = frag2(Q4[qrow4 + 4 + half], Q4[qrow4 + 6 + half]);

  float q2v[8], mrun[8], ssum[8];
  v8f o[4] = {v8f{}, v8f{}, v8f{}, v8f{}};
#pragma unroll
  for (int j = 0; j < 8; ++j) {
    q2v[j] = q2h[qrow0 + (half ? 8 + j : j)];
    mrun[j] = -3.0e38f;
    ssum[j] = 0.f;
  }

  const uint4* kt4 = (const uint4*)kt;
  const uint4* vt4 = (const uint4*)vt;
  const uint4* ps4 = (const uint4*)pst[w];

  for (int sb = 0; sb < SK / 32; ++sb) {
    const int s0 = sb * 32;
    __syncthreads();
    {
      // ---- K tile: straight 4KB copy (async DMA to LDS when available) ----
#if USE_ASYNC_LDS
      gv4i* gsrc = (gv4i*)(K32 + s0 * (DH / 2)) + tid * 2;
      lv4i* ldst = (lv4i*)kt + tid * 2;
      __builtin_amdgcn_global_load_async_to_lds_b128(gsrc + 0, ldst + 0, 0, 0);
      __builtin_amdgcn_global_load_async_to_lds_b128(gsrc + 1, ldst + 1, 0, 0);
#else
      const uint4* ks = (const uint4*)(K32 + s0 * (DH / 2));
      ((uint4*)kt)[tid * 2 + 0] = ks[tid * 2 + 0];
      ((uint4*)kt)[tid * 2 + 1] = ks[tid * 2 + 1];
#endif
      // ---- V tile: 16-bit transpose while staging ----
      const unsigned* vs = V32 + s0 * (DH / 2);
#pragma unroll
      for (int t = 0; t < 8; ++t) {
        const int i = tid * 8 + t;
        const unsigned u = vs[i];
        const int s = i >> 5, d0 = (i & 31) * 2;
        vt[d0 * 32 + s]       = (unsigned short)u;
        vt[(d0 + 1) * 32 + s] = (unsigned short)(u >> 16);
      }
      if (tid < 32) k2s[tid] = k2h[s0 + tid];
      if (sb + 1 < SK / 32)   // global_prefetch_b8 of next K tile
        __builtin_prefetch(K32 + (s0 + 32) * (DH / 2) + tid * 8, 0, 1);
    }
#if USE_ASYNC_LDS
    asm volatile("s_wait_asynccnt 0x0" ::: "memory");
#endif
    __syncthreads();

    // ---- scores = -||q-k|| / 8 for two 16-key column tiles ----
    v8f sc[2];
#pragma unroll
    for (int ct = 0; ct < 2; ++ct) {
      const int sl8 = (ct * 16 + l15) * (DH / 8);    // row base in uint4
      const v16bf blo = frag2(kt4[sl8 + half], kt4[sl8 + 2 + half]);
      const v16bf bhi = frag2(kt4[sl8 + 4 + half], kt4[sl8 + 6 + half]);
      v8f c = {};
      c = __builtin_amdgcn_wmma_f32_16x16x32_bf16(false, qa0, false, blo, (short)0,
                                                  c, false, false);
      c = __builtin_amdgcn_wmma_f32_16x16x32_bf16(false, qa1, false, bhi, (short)0,
                                                  c, false, false);
      const float kc2 = k2s[ct * 16 + l15];
#pragma unroll
      for (int j = 0; j < 8; ++j) {
        float d2 = fmaxf(q2v[j] + kc2 - 2.f * c[j], 0.f);
        c[j] = -sqrtf(d2) * 0.125f;
      }
      sc[ct] = c;
    }

    // ---- online softmax (row j lives across 16 lanes of each half-wave) ----
#pragma unroll
    for (int j = 0; j < 8; ++j) {
      float bm = fmaxf(sc[0][j], sc[1][j]);
#pragma unroll
      for (int msk = 1; msk < 16; msk <<= 1) bm = fmaxf(bm, __shfl_xor(bm, msk, 32));
      const float mnew = fmaxf(mrun[j], bm);
      const float corr = __expf(mrun[j] - mnew);
      mrun[j] = mnew;
      const float p0 = __expf(sc[0][j] - mnew);
      const float p1 = __expf(sc[1][j] - mnew);
      sc[0][j] = p0;
      sc[1][j] = p1;
      float ps = p0 + p1;
#pragma unroll
      for (int msk = 1; msk < 16; msk <<= 1) ps += __shfl_xor(ps, msk, 32);
      ssum[j] = ssum[j] * corr + ps;
#pragma unroll
      for (int dt = 0; dt < 4; ++dt) o[dt][j] *= corr;
    }

    // ---- P: C-layout -> LDS -> A-fragment layout (wave-private, LDS in-order) ----
#pragma unroll
    for (int ct = 0; ct < 2; ++ct)
#pragma unroll
      for (int j = 0; j < 8; ++j) {
        const int row = half ? 8 + j : j;
        pst[w][row * 32 + ct * 16 + l15] = f2bf(sc[ct][j]);
      }
    asm volatile("s_wait_dscnt 0x0" ::: "memory");
    const v16bf pa = frag2(ps4[l15 * 4 + half], ps4[l15 * 4 + 2 + half]);

    // ---- O += P @ V : four 16-wide d tiles, K = 32 keys ----
#pragma unroll
    for (int dt = 0; dt < 4; ++dt) {
      const int vr4 = (dt * 16 + l15) * 4;
      const v16bf vb = frag2(vt4[vr4 + half], vt4[vr4 + 2 + half]);
      o[dt] = __builtin_amdgcn_wmma_f32_16x16x32_bf16(false, pa, false, vb, (short)0,
                                                      o[dt], false, false);
    }
  }

  // ---- normalize, write ctx[n][l][e] with e = d*H + h ----
  const int n = nh >> 3, h = nh & 7;
#pragma unroll
  for (int j = 0; j < 8; ++j) {
    const int l = qrow0 + (half ? 8 + j : j);
    const float inv = 1.f / ssum[j];
    unsigned short* dst = ctx + ((size_t)n * LQ + l) * EE + h;
#pragma unroll
    for (int dt = 0; dt < 4; ++dt)
      dst[(size_t)(dt * 16 + l15) * HH] = f2bf(o[dt][j] * inv);
  }
}

// ---------------------------------------------------------------------------
// Kernel 4: result = ctx @ Wo^T, fp32 output
// ---------------------------------------------------------------------------
__global__ void __launch_bounds__(128) out_proj(
    const unsigned short* __restrict__ ctx, const unsigned short* __restrict__ Wob,
    float* __restrict__ out) {
  const int wid  = blockIdx.x * 4 + (threadIdx.x >> 5);
  const int lane = threadIdx.x & 31;
  const int half = lane >> 4, l15 = lane & 15;
  const int m0 = (wid / (EE / 16)) * 16;
  const int c0 = (wid % (EE / 16)) * 16;

  const uint4* A4 = (const uint4*)ctx;
  const uint4* B4 = (const uint4*)Wob;
  const int arow = (m0 + l15) * (EE / 8);
  const int brow = (c0 + l15) * (EE / 8);

  v8f acc = {};
  for (int kc = 0; kc < EE / 32; ++kc) {
    const int e4 = kc * 4;
    const uint4 alo = A4[arow + e4 + half];
    const uint4 ahi = A4[arow + e4 + 2 + half];
    const uint4 blo = B4[brow + e4 + half];
    const uint4 bhi = B4[brow + e4 + 2 + half];
    acc = __builtin_amdgcn_wmma_f32_16x16x32_bf16(false, frag2(alo, ahi), false,
                                                  frag2(blo, bhi), (short)0, acc,
                                                  false, false);
  }
#pragma unroll
  for (int j = 0; j < 8; ++j) {
    const int m = m0 + (half ? 8 + j : j);
    out[(size_t)m * EE + c0 + l15] = acc[j];
  }
}

// ---------------------------------------------------------------------------
extern "C" void kernel_launch(void* const* d_in, const int* in_sizes, int n_in,
                              void* d_out, int out_size, void* d_ws, size_t ws_size,
                              hipStream_t stream) {
  (void)in_sizes; (void)n_in; (void)out_size; (void)ws_size;
  const float* query = (const float*)d_in[0];
  const float* key   = (const float*)d_in[1];
  // d_in[2] = value: unused (reference faithfully projects `key` through Wv)
  const float* Wq = (const float*)d_in[3];
  const float* Wk = (const float*)d_in[4];
  const float* Wv = (const float*)d_in[5];
  const float* Wo = (const float*)d_in[6];
  float* out = (float*)d_out;

  char* ws = (char*)d_ws;
  const size_t MB = 1024 * 1024;
  unsigned short* Qbf = (unsigned short*)(ws + 0 * MB);    // 4 MiB each
  unsigned short* Kbf = (unsigned short*)(ws + 4 * MB);
  unsigned short* Vbf = (unsigned short*)(ws + 8 * MB);
  unsigned short* ctx = (unsigned short*)(ws + 12 * MB);
  float* q2 = (float*)(ws + 16 * MB);                      // 128 KiB
  float* k2 = (float*)(ws + 16 * MB + 256 * 1024);
  unsigned short* Xq_bf = (unsigned short*)(ws + 17 * MB); // 4 MiB
  unsigned short* Xk_bf = (unsigned short*)(ws + 21 * MB); // 4 MiB
  unsigned short* Wq_bf = (unsigned short*)(ws + 25 * MB); // 512 KiB each
  unsigned short* Wk_bf = (unsigned short*)(ws + 25 * MB + 512 * 1024);
  unsigned short* Wv_bf = (unsigned short*)(ws + 26 * MB);
  unsigned short* Wo_bf = (unsigned short*)(ws + 26 * MB + 512 * 1024);

  const int ndwX = NB * LQ * EE / 2;   // 1,048,576 dwords
  const int ndwW = EE * EE / 2;        // 131,072 dwords
  cvt_bf16<<<(ndwX + 255) / 256, 256, 0, stream>>>(query, (unsigned*)Xq_bf, ndwX);
  cvt_bf16<<<(ndwX + 255) / 256, 256, 0, stream>>>(key,   (unsigned*)Xk_bf, ndwX);
  cvt_bf16<<<(ndwW + 255) / 256, 256, 0, stream>>>(Wq, (unsigned*)Wq_bf, ndwW);
  cvt_bf16<<<(ndwW + 255) / 256, 256, 0, stream>>>(Wk, (unsigned*)Wk_bf, ndwW);
  cvt_bf16<<<(ndwW + 255) / 256, 256, 0, stream>>>(Wv, (unsigned*)Wv_bf, ndwW);
  cvt_bf16<<<(ndwW + 255) / 256, 256, 0, stream>>>(Wo, (unsigned*)Wo_bf, ndwW);

  proj_qkv<<<6144, 128, 0, stream>>>(Xq_bf, Xk_bf, Wq_bf, Wk_bf, Wv_bf, Qbf, Kbf, Vbf);
  row_norms<<<(NB * HH * LQ + 127) / 128, 128, 0, stream>>>(Qbf, Kbf, q2, k2);
  flash_attn<<<NB * HH * (LQ / 64), 128, 0, stream>>>(Qbf, Kbf, Vbf, q2, k2, ctx);
  out_proj<<<(NB * LQ / 16) * (EE / 16) / 4, 128, 0, stream>>>(ctx, Wo_bf, out);
}
